// Processor_27315992002795
// MI455X (gfx1250) — matrix operand order repaired and checked by reading, not compiled, not run on hardware
//
#include <hip/hip_runtime.h>
#include <hip/hip_bf16.h>

#define NN 50000
#define EE 600000
#define DD 128
#define NSTEPS 10

#define NWAVES 6                  // waves per block
#define ASE    392                // A stride (elements) for 384-wide inputs (784B, 16B-aligned)
#define ASN    264                // A stride for 256-wide inputs (528B, 16B-aligned)
#define AREG   (16 * ASE)         // 6272 ushorts per A buffer
#define HS     136                // hidden activation stride
#define H2OFF  2176               // second hidden buffer offset (ushorts)
#define WLDSO  (NWAVES * 2 * AREG)  // ushort offset of weights in dynamic LDS (75264)

typedef __attribute__((ext_vector_type(16))) __bf16 v16bf;
typedef __attribute__((ext_vector_type(8)))  float  v8f;

#define WAVE_LDS_FENCE()  asm volatile("s_wait_dscnt 0x0" ::: "memory")
#define WAIT_ASYNC0()     asm volatile("s_wait_asynccnt 0x0" ::: "memory")
#define WAIT_ASYNC24()    asm volatile("s_wait_asynccnt 0x18" ::: "memory")

static __device__ __forceinline__ unsigned short f2bs(float f) {
  union { float f; unsigned u; } v; v.f = f;
  unsigned u = v.u;
  u += 0x7fffu + ((u >> 16) & 1u);   // round-to-nearest-even bf16
  return (unsigned short)(u >> 16);
}

union FragU { v16bf v; uint4 q[2]; };

// 16B-granule swizzle for LDS weight rows: B-fragment reads hit each 4-bank group exactly twice
static __device__ __forceinline__ int wswz(int k, int g) {
  return g ^ (k & 15) ^ (((k >> 4) & 1) << 3);
}

// Async-stage a row-major bf16 weight matrix [rows][128] global -> LDS (swizzled granules)
static __device__ __forceinline__ void stage_weights(const unsigned short* __restrict__ gw,
                                                     unsigned short* lw, int rows, int tid) {
  unsigned ldsbase = (unsigned)(size_t)(void*)lw;
  const int chunks = rows * 16;      // 16B granules
  for (int c = tid; c < chunks; c += NWAVES * 32) {
    int k = c >> 4, g = c & 15;
    unsigned goff = (unsigned)c * 16u;
    unsigned loff = ldsbase + (unsigned)(k * 16 + wswz(k, g)) * 16u;
    asm volatile("global_load_async_to_lds_b128 %0, %1, %2"
                 :: "v"(loff), "v"(goff), "s"(gw) : "memory");
  }
}

// A fragment (16x32 bf16, MxK) from LDS activations
static __device__ __forceinline__ v16bf load_a_lds(const unsigned short* base, int stride,
                                                   int kc, int lane) {
  const int row = lane & 15;
  const int b8  = (lane < 16) ? 0 : 8;
  const unsigned short* p = base + row * stride + kc + b8;
  FragU u;
  u.q[0] = *(const uint4*)(p);
  u.q[1] = *(const uint4*)(p + 16);
  return u.v;
}

// B fragment (32x16 bf16, KxN) from swizzled LDS weights: lane%16 = N column, elements over K in table order
static __device__ __forceinline__ v16bf load_b_lds(const unsigned short* Wl, int kc, int t, int lane) {
  const int n = t * 16 + (lane & 15);
  const int h8 = (lane < 16) ? 0 : 8;
  union { v16bf v; unsigned short h[16]; } u;
#pragma unroll
  for (int s = 0; s < 16; ++s) {
    const int k = kc + h8 + s + ((s >= 8) ? 8 : 0);
    u.h[s] = Wl[k * DD + wswz(k, n >> 3) * 8 + (n & 7)];
  }
  return u.v;
}

// [16 x 32*KC] @ [32*KC x 128] + bias (+ReLU); acc[t][r] = (M=r+8*(lane>=16), N=t*16+lane%16)
template<int KC, bool RELU>
static __device__ __forceinline__ void gemm_layer(const unsigned short* abase, int astride,
                                                  const unsigned short* Wl,
                                                  const float* __restrict__ bias,
                                                  int lane, v8f acc[8]) {
#pragma unroll
  for (int t = 0; t < 8; ++t)
#pragma unroll
    for (int r = 0; r < 8; ++r) acc[t][r] = 0.f;

  for (int k = 0; k < KC; ++k) {
    v16bf a = load_a_lds(abase, astride, k * 32, lane);
    v16bf b[8];
#pragma unroll
    for (int t = 0; t < 8; ++t) b[t] = load_b_lds(Wl, k * 32, t, lane);
#pragma unroll
    for (int t = 0; t < 8; ++t)
      acc[t] = __builtin_amdgcn_wmma_f32_16x16x32_bf16(false, a, false, b[t],
                                                       (short)0, acc[t], false, false);
  }
  const int nlo = lane & 15;
#pragma unroll
  for (int t = 0; t < 8; ++t) {
    float bv = bias[t * 16 + nlo];
#pragma unroll
    for (int r = 0; r < 8; ++r) {
      float v = acc[t][r] + bv;
      if (RELU) v = fmaxf(v, 0.f);
      acc[t][r] = v;
    }
  }
}

static __device__ __forceinline__ void store_act(unsigned short* dst, int stride, int lane,
                                                 const v8f acc[8]) {
  const int mbase = (lane < 16) ? 0 : 8;
  const int nlo = lane & 15;
#pragma unroll
  for (int t = 0; t < 8; ++t)
#pragma unroll
    for (int r = 0; r < 8; ++r)
      dst[(mbase + r) * stride + t * 16 + nlo] = f2bs(acc[t][r]);
}

// LayerNorm over N=128 per row; half-wave reduction with shfl_xor
static __device__ __forceinline__ void lnorm(v8f acc[8], const float* __restrict__ g,
                                             const float* __restrict__ beta, int lane) {
  float s[8], q[8];
#pragma unroll
  for (int r = 0; r < 8; ++r) {
    float a = 0.f, b = 0.f;
#pragma unroll
    for (int t = 0; t < 8; ++t) { float v = acc[t][r]; a += v; b += v * v; }
    s[r] = a; q[r] = b;
  }
#pragma unroll
  for (int m = 1; m < 16; m <<= 1) {
#pragma unroll
    for (int r = 0; r < 8; ++r) {
      s[r] += __shfl_xor(s[r], m, 32);
      q[r] += __shfl_xor(q[r], m, 32);
    }
  }
#pragma unroll
  for (int r = 0; r < 8; ++r) {
    float mu  = s[r] * (1.f / 128.f);
    float var = q[r] * (1.f / 128.f) - mu * mu;
    q[r] = rsqrtf(var + 1e-5f);
    s[r] = mu;
  }
  const int nlo = lane & 15;
#pragma unroll
  for (int t = 0; t < 8; ++t) {
    float gv = g[t * 16 + nlo], bv = beta[t * 16 + nlo];
#pragma unroll
    for (int r = 0; r < 8; ++r)
      acc[t][r] = (acc[t][r] - s[r]) * q[r] * gv + bv;
  }
}

// Async gather of one 16x384 edge-input tile: [xg[i0[e]], xg[i1[e]], ebase[e]]
static __device__ __forceinline__ void edge_gather_async(
    const unsigned short* __restrict__ xg, const unsigned short* __restrict__ ebase,
    const int* __restrict__ i0, const int* __restrict__ i1,
    int e0, unsigned AbaseB, int lane) {
#pragma unroll
  for (int seg = 0; seg < 3; ++seg) {
    for (int i = lane; i < 256; i += 32) {
      int row = i >> 4, ch = i & 15;
      int e = e0 + row;
      const unsigned short* sp;
      if (seg == 2)      sp = ebase + (size_t)e * DD;
      else if (seg == 0) sp = xg + (size_t)i0[e] * DD;
      else               sp = xg + (size_t)i1[e] * DD;
      unsigned loff = AbaseB + (unsigned)((row * ASE + seg * DD + ch * 8) * 2);
      asm volatile("global_load_async_to_lds_b128 %0, %1, off"
                   :: "v"(loff), "v"(sp + ch * 8) : "memory");
    }
  }
}

// 3-layer MLP + LayerNorm on one staged 16-row tile
template<int KC0>
static __device__ __forceinline__ void mlp3(unsigned short* A, int as0,
    const unsigned short* WL0, const unsigned short* WL1, const unsigned short* WL2,
    const float* b0, const float* b1, const float* b2,
    const float* g, const float* beta, int lane, v8f acc[8]) {
  gemm_layer<KC0, true>(A, as0, WL0, b0, lane, acc);
  store_act(A, HS, lane, acc);
  WAVE_LDS_FENCE();
  gemm_layer<4, true>(A, HS, WL1, b1, lane, acc);
  store_act(A + H2OFF, HS, lane, acc);
  WAVE_LDS_FENCE();
  gemm_layer<4, false>(A + H2OFF, HS, WL2, b2, lane, acc);
  lnorm(acc, g, beta, lane);
}

// ---------------- edge message: m = MLP_LN([x[tgt], x[src], e]); aggr[tgt] += m ----------------
__global__ __launch_bounds__(NWAVES * 32) void edge_msg_kernel(
    const unsigned short* __restrict__ xb, const unsigned short* __restrict__ eb,
    const int* __restrict__ ei,
    const unsigned short* __restrict__ W0, const unsigned short* __restrict__ W1,
    const unsigned short* __restrict__ W2,
    const float* __restrict__ b0, const float* __restrict__ b1, const float* __restrict__ b2,
    const float* __restrict__ g, const float* __restrict__ beta,
    float* __restrict__ aggr) {
  extern __shared__ unsigned short dynsm[];
  const int tid = threadIdx.x, wave = tid >> 5, lane = tid & 31;

  unsigned short* WL0 = dynsm + WLDSO;
  unsigned short* WL1 = WL0 + 384 * DD;
  unsigned short* WL2 = WL1 + DD * DD;
  stage_weights(W0, WL0, 384, tid);
  stage_weights(W1, WL1, DD, tid);
  stage_weights(W2, WL2, DD, tid);
  WAIT_ASYNC0();
  __syncthreads();

  unsigned short* buf0 = dynsm + wave * (2 * AREG);
  unsigned short* buf1 = buf0 + AREG;
  const unsigned ab0 = (unsigned)(size_t)(void*)buf0;
  const unsigned ab1 = (unsigned)(size_t)(void*)buf1;

  const int TILES = EE / 16;
  const int tstr = gridDim.x * NWAVES;
  int tile = blockIdx.x * NWAVES + wave;
  if (tile < TILES) edge_gather_async(xb, eb, ei + EE, ei, tile * 16, ab0, lane);

  int sel = 0;
  for (; tile < TILES; tile += tstr, sel ^= 1) {
    unsigned short* A = sel ? buf1 : buf0;
    const int next = tile + tstr;
    if (next < TILES) {                 // prefetch next tile into the other buffer
      WAVE_LDS_FENCE();
      edge_gather_async(xb, eb, ei + EE, ei, next * 16, sel ? ab0 : ab1, lane);
      WAIT_ASYNC24();                   // in-order: current tile's 24 asyncs complete
    } else {
      WAIT_ASYNC0();
    }

    v8f acc[8];
    mlp3<12>(A, ASE, WL0, WL1, WL2, b0, b1, b2, g, beta, lane, acc);

    const int e0 = tile * 16;
    const int mbase = (lane < 16) ? 0 : 8, nlo = lane & 15;
#pragma unroll
    for (int r = 0; r < 8; ++r) {
      int e = e0 + mbase + r;
      float* dst = aggr + (size_t)ei[EE + e] * DD + nlo;
#pragma unroll
      for (int t = 0; t < 8; ++t) atomicAdd(dst + t * 16, acc[t][r]);
    }
  }
}

// ---------------- node update: x_upd = MLP_LN([aggr, x]); x += x_upd ----------------
__global__ __launch_bounds__(NWAVES * 32) void node_kernel(
    const float* __restrict__ aggr,
    unsigned short* __restrict__ xb,    // bf16 x (in: old, out: new)
    float* __restrict__ xf,             // f32 x in d_out (in: old, out: new)
    unsigned short* __restrict__ xub,   // bf16 x_upd (out)
    const unsigned short* __restrict__ W0, const unsigned short* __restrict__ W1,
    const unsigned short* __restrict__ W2,
    const float* __restrict__ b0, const float* __restrict__ b1, const float* __restrict__ b2,
    const float* __restrict__ g, const float* __restrict__ beta) {
  extern __shared__ unsigned short dynsm[];
  const int tid = threadIdx.x, wave = tid >> 5, lane = tid & 31;

  unsigned short* WL0 = dynsm + WLDSO;
  unsigned short* WL1 = WL0 + 256 * DD;
  unsigned short* WL2 = WL1 + DD * DD;
  stage_weights(W0, WL0, 256, tid);
  stage_weights(W1, WL1, DD, tid);
  stage_weights(W2, WL2, DD, tid);
  WAIT_ASYNC0();
  __syncthreads();

  unsigned short* A = dynsm + wave * (2 * AREG);
  const unsigned Abase = (unsigned)(size_t)(void*)A;

  for (int tile = blockIdx.x * NWAVES + wave; tile < NN / 16; tile += gridDim.x * NWAVES) {
    const int n0 = tile * 16;
    WAVE_LDS_FENCE();
    for (int i = lane; i < 512; i += 32) {           // seg0: aggr f32 -> bf16 via VGPRs
      int row = i >> 5, c = i & 31;
      float4 v = *(const float4*)(aggr + (size_t)(n0 + row) * DD + c * 4);
      union { unsigned short h[4]; uint2 u; } pk;
      pk.h[0] = f2bs(v.x); pk.h[1] = f2bs(v.y); pk.h[2] = f2bs(v.z); pk.h[3] = f2bs(v.w);
      *(uint2*)(A + row * ASN + c * 4) = pk.u;
    }
    for (int i = lane; i < 256; i += 32) {           // seg1: x bf16 async copy
      int row = i >> 4, ch = i & 15;
      const unsigned short* sp = xb + (size_t)(n0 + row) * DD + ch * 8;
      unsigned loff = Abase + (unsigned)((row * ASN + DD + ch * 8) * 2);
      asm volatile("global_load_async_to_lds_b128 %0, %1, off"
                   :: "v"(loff), "v"(sp) : "memory");
    }
    WAVE_LDS_FENCE();
    WAIT_ASYNC0();

    v8f acc[8];
    mlp3<8>(A, ASN, WL0, WL1, WL2, b0, b1, b2, g, beta, lane, acc);

    const int mbase = (lane < 16) ? 0 : 8, nlo = lane & 15;
#pragma unroll
    for (int r = 0; r < 8; ++r) {
      size_t row = (size_t)(n0 + mbase + r) * DD + nlo;
#pragma unroll
      for (int t = 0; t < 8; ++t) {
        float u = acc[t][r];
        xub[row + t * 16] = f2bs(u);                 // pre-residual update (for edge_update)
        float xn = u + xf[row + t * 16];             // residual
        xf[row + t * 16] = xn;
        xb[row + t * 16] = f2bs(xn);
      }
    }
  }
}

// ---------------- edge update: e_upd = MLP_LN([x_upd[src], x_upd[tgt], e]); e += e_upd ----------------
__global__ __launch_bounds__(NWAVES * 32) void edge_upd_kernel(
    const unsigned short* __restrict__ xub,
    unsigned short* __restrict__ eb,    // bf16 e (in: old, out: new)
    float* __restrict__ ef,             // f32 e in d_out (in: old, out: new)
    const int* __restrict__ ei,
    const unsigned short* __restrict__ W0, const unsigned short* __restrict__ W1,
    const unsigned short* __restrict__ W2,
    const float* __restrict__ b0, const float* __restrict__ b1, const float* __restrict__ b2,
    const float* __restrict__ g, const float* __restrict__ beta) {
  extern __shared__ unsigned short dynsm[];
  const int tid = threadIdx.x, wave = tid >> 5, lane = tid & 31;

  unsigned short* WL0 = dynsm + WLDSO;
  unsigned short* WL1 = WL0 + 384 * DD;
  unsigned short* WL2 = WL1 + DD * DD;
  stage_weights(W0, WL0, 384, tid);
  stage_weights(W1, WL1, DD, tid);
  stage_weights(W2, WL2, DD, tid);
  WAIT_ASYNC0();
  __syncthreads();

  unsigned short* buf0 = dynsm + wave * (2 * AREG);
  unsigned short* buf1 = buf0 + AREG;
  const unsigned ab0 = (unsigned)(size_t)(void*)buf0;
  const unsigned ab1 = (unsigned)(size_t)(void*)buf1;

  const int TILES = EE / 16;
  const int tstr = gridDim.x * NWAVES;
  int tile = blockIdx.x * NWAVES + wave;
  if (tile < TILES) edge_gather_async(xub, eb, ei, ei + EE, tile * 16, ab0, lane);

  int sel = 0;
  for (; tile < TILES; tile += tstr, sel ^= 1) {
    unsigned short* A = sel ? buf1 : buf0;
    const int next = tile + tstr;
    if (next < TILES) {
      WAVE_LDS_FENCE();
      edge_gather_async(xub, eb, ei, ei + EE, next * 16, sel ? ab0 : ab1, lane);
      WAIT_ASYNC24();
    } else {
      WAIT_ASYNC0();
    }

    v8f acc[8];
    mlp3<12>(A, ASE, WL0, WL1, WL2, b0, b1, b2, g, beta, lane, acc);

    const int e0 = tile * 16;
    const int mbase = (lane < 16) ? 0 : 8, nlo = lane & 15;
#pragma unroll
    for (int r = 0; r < 8; ++r) {
      size_t row = (size_t)(e0 + mbase + r) * DD + nlo;
#pragma unroll
      for (int t = 0; t < 8; ++t) {
        float en = acc[t][r] + ef[row + t * 16];     // residual
        ef[row + t * 16] = en;
        eb[row + t * 16] = f2bs(en);
      }
    }
  }
}

// ---------------- helpers ----------------
__global__ void cvt_bf16_kernel(const float* __restrict__ src, unsigned short* __restrict__ dst, int n) {
  int i = blockIdx.x * 256 + threadIdx.x;
  if (i < n) dst[i] = f2bs(src[i]);
}

__global__ void init_copy_kernel(const float* __restrict__ src, float* __restrict__ dstf,
                                 unsigned short* __restrict__ dstb, int n) {
  int i = blockIdx.x * 256 + threadIdx.x;
  if (i < n) { float v = src[i]; dstf[i] = v; dstb[i] = f2bs(v); }
}

extern "C" void kernel_launch(void* const* d_in, const int* in_sizes, int n_in,
                              void* d_out, int out_size, void* d_ws, size_t ws_size,
                              hipStream_t stream) {
  const float* x_in = (const float*)d_in[0];
  const int*   ei   = (const int*)d_in[1];       // [2][E]: row0=src, row1=tgt (int32)
  const float* e_in = (const float*)d_in[2];

  float* out_x = (float*)d_out;                  // working x (f32)
  float* out_e = out_x + (size_t)NN * DD;        // working edge features (f32)

  char* ws = (char*)d_ws;
  unsigned short* xb   = (unsigned short*)(ws);                // bf16 x          12.8 MB
  unsigned short* xub  = (unsigned short*)(ws + 12800000);     // bf16 x_upd      12.8 MB
  unsigned short* eb   = (unsigned short*)(ws + 25600000);     // bf16 e         153.6 MB
  float*          aggr = (float*)(ws + 179200000);             // f32 aggr        25.6 MB
  unsigned short* wb   = (unsigned short*)(ws + 204800000);    // bf16 weights     4.6 MB

  const int EDGE_LDS = (WLDSO + 384 * DD + 2 * DD * DD) * 2;   // 314,368 B
  const int NODE_LDS = (WLDSO + 256 * DD + 2 * DD * DD) * 2;   // 281,600 B
  hipFuncSetAttribute(reinterpret_cast<const void*>(edge_msg_kernel),
                      hipFuncAttributeMaxDynamicSharedMemorySize, EDGE_LDS);
  hipFuncSetAttribute(reinterpret_cast<const void*>(node_kernel),
                      hipFuncAttributeMaxDynamicSharedMemorySize, NODE_LDS);
  hipFuncSetAttribute(reinterpret_cast<const void*>(edge_upd_kernel),
                      hipFuncAttributeMaxDynamicSharedMemorySize, EDGE_LDS);

  { int n = NN * DD; init_copy_kernel<<<(n + 255) / 256, 256, 0, stream>>>(x_in, out_x, xb, n); }
  { int n = EE * DD; init_copy_kernel<<<(n + 255) / 256, 256, 0, stream>>>(e_in, out_e, eb, n); }

  // params: d_in[3 + s*24 + mlp*8 + j]; mlp: 0=edge_fn 1=node_fn 2=edge_update_fn
  // j: 0..2 = w0..w2, 3..5 = b0..b2, 6 = g, 7 = beta
  static const int wsz[3][3] = {{384 * DD, DD * DD, DD * DD},
                                {256 * DD, DD * DD, DD * DD},
                                {384 * DD, DD * DD, DD * DD}};
  size_t wofs[NSTEPS][3][3];
  size_t wo = 0;
  for (int s = 0; s < NSTEPS; ++s)
    for (int m = 0; m < 3; ++m)
      for (int j = 0; j < 3; ++j) {
        int n = wsz[m][j];
        wofs[s][m][j] = wo;
        const float* src = (const float*)d_in[3 + s * 24 + m * 8 + j];
        cvt_bf16_kernel<<<(n + 255) / 256, 256, 0, stream>>>(src, wb + wo, n);
        wo += n;
      }

  const int edge_blocks = 512;                               // grid-stride over 37500 tiles
  const int node_blocks = (NN / 16 + NWAVES - 1) / NWAVES;   // 521, ~1 tile per wave

  for (int s = 0; s < NSTEPS; ++s) {
    auto B = [&](int m, int j) { return (const float*)d_in[3 + s * 24 + m * 8 + j]; };

    hipMemsetAsync(aggr, 0, (size_t)NN * DD * sizeof(float), stream);

    edge_msg_kernel<<<edge_blocks, NWAVES * 32, EDGE_LDS, stream>>>(
        xb, eb, ei,
        wb + wofs[s][0][0], wb + wofs[s][0][1], wb + wofs[s][0][2],
        B(0, 3), B(0, 4), B(0, 5), B(0, 6), B(0, 7), aggr);

    node_kernel<<<node_blocks, NWAVES * 32, NODE_LDS, stream>>>(
        aggr, xb, out_x, xub,
        wb + wofs[s][1][0], wb + wofs[s][1][1], wb + wofs[s][1][2],
        B(1, 3), B(1, 4), B(1, 5), B(1, 6), B(1, 7));

    edge_upd_kernel<<<edge_blocks, NWAVES * 32, EDGE_LDS, stream>>>(
        xub, eb, out_e, ei,
        wb + wofs[s][2][0], wb + wofs[s][2][1], wb + wofs[s][2][2],
        B(2, 3), B(2, 4), B(2, 5), B(2, 6), B(2, 7));
  }
}
